// FullyConnectedTensorProduct_15461882266097
// MI455X (gfx1250) — matrix operand order repaired and checked by reading, not compiled, not run on hardware
//
#include <hip/hip_runtime.h>

// ---------------------------------------------------------------------------
// CDNA5 (gfx1250) fused e3nn FullyConnectedTensorProduct
//   o0  = A000*(s1 (x) s2) @ w000 + A110*(sum_i a_i (x) b_i) @ w110
//   o1k = A011*(s1 (x) b_k) @ w011 + A101*(a_k (x) s2) @ w101
//   o2k = A111*(a_{k+1}(x)b_{k+2} - a_{k+2}(x)b_{k+1}) @ w111
// All contractions run on v_wmma_f32_16x16x32_bf16 (f32 accumulate).
// Wave split: q=0 waves own {o0,o2*} (5 A-fragments), q=1 waves own {o1*}
// (6 A-fragments); each wave covers the full 64-column width so no
// A-fragment is ever built twice inside a workgroup.
// ---------------------------------------------------------------------------

typedef __attribute__((ext_vector_type(16))) __bf16 v16bf;
typedef __attribute__((ext_vector_type(8)))  float  v8f;

#define BATCH   32768
#define MULIN   32
#define DIMIN   128
#define DIMOUT  448
#define MTILE   64           // batch rows per workgroup
#define NFRag   512          // bf16 elements per 32x16 B-fragment

// A000 = A011 = A101 = A111 = 1/sqrt(2048); A110 = 1/sqrt(6144)
#define SCALE_MAIN 0.022097086912079608f
#define SCALE_110  0.012757759747234188f

// float -> bf16 (round to nearest even), host-type-free
__device__ __forceinline__ unsigned short f2bf(float x) {
  unsigned int u = __float_as_uint(x);
  unsigned int r = (u + 0x7fffu + ((u >> 16) & 1u)) >> 16;
  return (unsigned short)r;
}

__device__ __forceinline__ v16bf cvt16(const float f[16]) {
  v16bf r;
#pragma unroll
  for (int i = 0; i < 16; ++i) r[i] = (__bf16)f[i];
  return r;
}

__device__ __forceinline__ v8f wmma_bf16(v16bf a, v16bf b, v8f c) {
  // (neg_a, A, neg_b, B, c_mod, C, reuse_a, reuse_b)
  return __builtin_amdgcn_wmma_f32_16x16x32_bf16(false, a, false, b,
                                                 (short)0, c, false, false);
}

// ---------------------------------------------------------------------------
// Prep: f32 weights -> bf16 WMMA B-fragments in d_ws (scales baked in).
// Fragment (t, u, ntile) is a 32(K=v) x 16(N=w) tile stored lane-major:
//   element idx = ((((t*32+u)*4+ntile)*32 + lane)*8 + r)*2 + h
//   n = ntile*16 + lane%16 ;  k = (lane/16)*8 + 2r + h + (r >= 4 ? 8 : 0)
// (B layout assumption: lanes 0-15 carry K=0..7,16..23, lanes 16-31 K=8..15,24..31,
//  two bf16 K-values packed per VGPR — mirrors ISA 7.12.5 B conventions.)
// ---------------------------------------------------------------------------
__global__ void __launch_bounds__(256) prep_weights(
    const float* __restrict__ w000, const float* __restrict__ w110,
    const float* __restrict__ w011, const float* __restrict__ w101,
    const float* __restrict__ w111, unsigned short* __restrict__ wsu) {
  const int idx = blockIdx.x * 256 + threadIdx.x;     // 0 .. 327679
  const int h     = idx & 1;
  const int r     = (idx >> 1) & 7;
  const int lane  = (idx >> 4) & 31;
  const int ntile = (idx >> 9) & 3;
  const int u     = (idx >> 11) & 31;
  const int t     = idx >> 16;                        // 0..4

  const float* W = (t == 0) ? w000 : (t == 1) ? w110
                 : (t == 2) ? w011 : (t == 3) ? w101 : w111;
  const float sc = (t == 1) ? SCALE_110 : SCALE_MAIN;

  const int n = ntile * 16 + (lane & 15);
  const int k = ((lane >> 4) << 3) + 2 * r + h + (r >= 4 ? 8 : 0);
  const float val = W[(size_t)(u * 32 + k) * 64 + n] * sc;
  wsu[idx] = f2bf(val);
}

// ---------------------------------------------------------------------------
// Main fused kernel. 256 threads = 8 waves; wave = (mw = wid%4) 16-row tile
// x (q = wid/4) output-group set. One u-step == one K=32 WMMA chunk (all v).
// ---------------------------------------------------------------------------
__global__ void __launch_bounds__(256) tp_kernel(
    const float* __restrict__ x1, const float* __restrict__ x2,
    const unsigned short* __restrict__ wsu, float* __restrict__ out) {
  __shared__ float lds_x1[MTILE * DIMIN];   // repacked: [row][u*4 + {s,a0,a1,a2}]

  const int tid   = threadIdx.x;
  const int lane  = tid & 31;
  const int wid   = tid >> 5;
  const int mw    = wid & 3;
  const int q     = wid >> 2;               // 0: {o0,o2*}  1: {o1*}
  const int hi    = (lane >> 4) & 1;        // K-half ownership per A layout
  const int rowWG = blockIdx.x * MTILE;

  // ---- stage x1 tile into LDS, repacked so each u is one aligned float4 ----
#pragma unroll
  for (int it = 0; it < (MTILE * DIMIN) / 256; ++it) {
    const int e = it * 256 + tid;
    const int row = e >> 7, col = e & 127;
    const float v = x1[(size_t)(rowWG + row) * DIMIN + col];
    int dst;
    if (col < 32) dst = col * 4;                         // s1[u] -> slot 0
    else { const int cc = col - 32; dst = (cc / 3) * 4 + 1 + (cc % 3); }
    lds_x1[row * DIMIN + dst] = v;
  }
  __syncthreads();

  // ---- per-lane x2 preload in the exact A-fragment K order ----
  // lanes 0-15: v in {0..7,16..23}; lanes 16-31: v in {8..15,24..31}
  const int mloc = mw * 16 + (lane & 15);
  const float* x2row = x2 + (size_t)(rowWG + mloc) * DIMIN;
  float s2f[16], vb[3][16];
#pragma unroll
  for (int i = 0; i < 16; ++i) {
    const int v = (i < 8 ? i : i + 8) + hi * 8;
    s2f[i]   = x2row[v];
    vb[0][i] = x2row[32 + 3 * v + 0];
    vb[1][i] = x2row[32 + 3 * v + 1];
    vb[2][i] = x2row[32 + 3 * v + 2];
  }

  auto ldw = [&](int t, int nt, int uu) -> v16bf {
    const size_t off = ((((size_t)t * 32 + (size_t)uu) * 4 +
                         (size_t)nt) * NFRag) + (size_t)lane * 16;
    return *(const v16bf*)(wsu + off);
  };

  const v8f vzero = {0.f, 0.f, 0.f, 0.f, 0.f, 0.f, 0.f, 0.f};
  const int nlocal = lane & 15;
  const size_t rbase = (size_t)(rowWG + mw * 16 + 8 * hi);
  float f[16];
  v16bf af;

  if (q == 0) {
    // =====================================================================
    // q=0: o0 (w000,w110) and o2k (w111): 5 A-fragments, 20 WMMAs per u
    // =====================================================================
    v8f acc0[4], acc2[3][4];
#pragma unroll
    for (int nt = 0; nt < 4; ++nt) acc0[nt] = vzero;
#pragma unroll
    for (int k = 0; k < 3; ++k)
#pragma unroll
      for (int nt = 0; nt < 4; ++nt) acc2[k][nt] = vzero;

    for (int u = 0; u < MULIN; ++u) {
      if (u + 1 < MULIN) {
        __builtin_prefetch(wsu + (((size_t)0 * 32 + u + 1) * 4) * NFRag + lane * 16, 0, 1);
        __builtin_prefetch(wsu + (((size_t)1 * 32 + u + 1) * 4) * NFRag + lane * 16, 0, 1);
        __builtin_prefetch(wsu + (((size_t)4 * 32 + u + 1) * 4) * NFRag + lane * 16, 0, 1);
      }
      const float4 sa = *(const float4*)&lds_x1[mloc * DIMIN + u * 4];
      const float s1u = sa.x;
      const float aa[3] = {sa.y, sa.z, sa.w};

      // z000 = s1u * s2
#pragma unroll
      for (int i = 0; i < 16; ++i) f[i] = s1u * s2f[i];
      af = cvt16(f);
#pragma unroll
      for (int nt = 0; nt < 4; ++nt)
        acc0[nt] = wmma_bf16(af, ldw(0, nt, u), acc0[nt]);

      // z110 = sum_i a_i * b_i
#pragma unroll
      for (int i = 0; i < 16; ++i)
        f[i] = aa[0] * vb[0][i] + aa[1] * vb[1][i] + aa[2] * vb[2][i];
      af = cvt16(f);
#pragma unroll
      for (int nt = 0; nt < 4; ++nt)
        acc0[nt] = wmma_bf16(af, ldw(1, nt, u), acc0[nt]);

      // t_k = a_{k+1} b_{k+2} - a_{k+2} b_{k+1}
#pragma unroll
      for (int k = 0; k < 3; ++k) {
        const int k1 = (k + 1) % 3, k2 = (k + 2) % 3;
#pragma unroll
        for (int i = 0; i < 16; ++i)
          f[i] = aa[k1] * vb[k2][i] - aa[k2] * vb[k1][i];
        af = cvt16(f);
#pragma unroll
        for (int nt = 0; nt < 4; ++nt)
          acc2[k][nt] = wmma_bf16(af, ldw(4, nt, u), acc2[k][nt]);
      }
    }

    // ---- epilogue: o0 cols [0,64), o2 cols 256 + n*3 + k ----
#pragma unroll
    for (int nt = 0; nt < 4; ++nt) {
      const int n = nt * 16 + nlocal;
      float* op = out + rbase * DIMOUT + n;
#pragma unroll
      for (int j = 0; j < 8; ++j) op[(size_t)j * DIMOUT] = acc0[nt][j];
    }
#pragma unroll
    for (int k = 0; k < 3; ++k)
#pragma unroll
      for (int nt = 0; nt < 4; ++nt) {
        const int n = nt * 16 + nlocal;
        float* op = out + rbase * DIMOUT + 256 + n * 3 + k;
#pragma unroll
        for (int j = 0; j < 8; ++j) op[(size_t)j * DIMOUT] = acc2[k][nt][j];
      }
  } else {
    // =====================================================================
    // q=1: o1k (w011,w101): 6 A-fragments, 24 WMMAs per u
    // =====================================================================
    v8f acc1[3][4];
#pragma unroll
    for (int k = 0; k < 3; ++k)
#pragma unroll
      for (int nt = 0; nt < 4; ++nt) acc1[k][nt] = vzero;

    for (int u = 0; u < MULIN; ++u) {
      if (u + 1 < MULIN) {
        __builtin_prefetch(wsu + (((size_t)2 * 32 + u + 1) * 4) * NFRag + lane * 16, 0, 1);
        __builtin_prefetch(wsu + (((size_t)3 * 32 + u + 1) * 4) * NFRag + lane * 16, 0, 1);
      }
      const float4 sa = *(const float4*)&lds_x1[mloc * DIMIN + u * 4];
      const float s1u = sa.x;
      const float aa[3] = {sa.y, sa.z, sa.w};

#pragma unroll
      for (int k = 0; k < 3; ++k) {
        // z011k = s1u * b_k
#pragma unroll
        for (int i = 0; i < 16; ++i) f[i] = s1u * vb[k][i];
        af = cvt16(f);
#pragma unroll
        for (int nt = 0; nt < 4; ++nt)
          acc1[k][nt] = wmma_bf16(af, ldw(2, nt, u), acc1[k][nt]);

        // z101k = a_k * s2
#pragma unroll
        for (int i = 0; i < 16; ++i) f[i] = aa[k] * s2f[i];
        af = cvt16(f);
#pragma unroll
        for (int nt = 0; nt < 4; ++nt)
          acc1[k][nt] = wmma_bf16(af, ldw(3, nt, u), acc1[k][nt]);
      }
    }

    // ---- epilogue: o1 cols 64 + n*3 + k ----
#pragma unroll
    for (int k = 0; k < 3; ++k)
#pragma unroll
      for (int nt = 0; nt < 4; ++nt) {
        const int n = nt * 16 + nlocal;
        float* op = out + rbase * DIMOUT + 64 + n * 3 + k;
#pragma unroll
        for (int j = 0; j < 8; ++j) op[(size_t)j * DIMOUT] = acc1[k][nt][j];
      }
  }
}

// ---------------------------------------------------------------------------
extern "C" void kernel_launch(void* const* d_in, const int* in_sizes, int n_in,
                              void* d_out, int out_size, void* d_ws, size_t ws_size,
                              hipStream_t stream) {
  const float* x1   = (const float*)d_in[0];
  const float* x2   = (const float*)d_in[1];
  const float* w000 = (const float*)d_in[2];
  const float* w110 = (const float*)d_in[3];
  const float* w011 = (const float*)d_in[4];
  const float* w101 = (const float*)d_in[5];
  const float* w111 = (const float*)d_in[6];
  unsigned short* wsu = (unsigned short*)d_ws;   // 5*32*4*512 bf16 = 640 KB

  prep_weights<<<(5 * 32 * 4 * NFRag) / 256, 256, 0, stream>>>(
      w000, w110, w011, w101, w111, wsu);
  tp_kernel<<<BATCH / MTILE, 256, 0, stream>>>(
      x1, x2, wsu, (float*)d_out);
}
